// EncodeProcessDecode_35905926595217
// MI455X (gfx1250) — matrix operand order repaired and checked by reading, not compiled, not run on hardware
//
#include <hip/hip_runtime.h>

// ---------------------------------------------------------------------------
// MeshGraphNets-style encode/process/decode for MI455X (gfx1250, wave32).
// All GEMM layers run on V_WMMA_F32_16X16X32_F16 (f16 inputs, f32 accum).
// Layer-2/3 weights staged to LDS via global_load_async_to_lds_b128; the
// s_wait_asynccnt is deferred until after layer 1 so staging overlaps compute.
// ---------------------------------------------------------------------------

typedef _Float16 f16;
typedef __attribute__((ext_vector_type(16))) _Float16 v16h;
typedef __attribute__((ext_vector_type(8)))  float    v8f;

#define DEV static __device__ __forceinline__

union AFrag { v16h v; uint4 q[2]; };

DEV v8f wmma_f16(v16h a, v16h b, v8f c) {
  // D = A(16x32) * B(32x16) + C(16x16), f32 accumulate
  return __builtin_amdgcn_wmma_f32_16x16x32_f16(false, a, false, b, (short)0, c,
                                                false, false);
}

// A fragment (16x32 f16), ISA layout: lane<16 holds K octets {0..7,16..23},
// lane>=16 holds {8..15,24..31}; row M = lane&15. rowp = row base pointer.
DEV v16h a_frag(const f16* rowp, int kbase, int hf) {
  AFrag u;
  u.q[0] = *(const uint4*)(rowp + kbase + 8 * hf);
  u.q[1] = *(const uint4*)(rowp + kbase + 8 * hf + 16);
  return u.v;
}

// B fragment (32x16 f16) pre-packed: frag is 512 contiguous f16,
// lane l owns elements [l*16, l*16+16): N = l&15, K = 8*(l>>4)+16*(j>>3)+(j&7).
DEV v16h b_frag(const f16* frags, int fragIdx, int lane) {
  AFrag u;
  const uint4* p = (const uint4*)(frags + ((size_t)fragIdx << 9) + lane * 16);
  u.q[0] = p[0];
  u.q[1] = p[1];
  return u.v;
}

DEV void zero_acc(v8f (&acc)[2][8]) {
  v8f z = {};
#pragma unroll
  for (int s = 0; s < 2; ++s)
#pragma unroll
    for (int n = 0; n < 8; ++n) acc[s][n] = z;
}

// Issue block-wide async stage of weight fragments into LDS (ASYNCcnt path).
// NOTE: no wait here -- call stage_wait() when the data is actually needed.
DEV void stage_issue(f16* ldsDst, const f16* src, int elems) {
  int chunks = elems >> 3;  // 16B per chunk
  for (int c = threadIdx.x; c < chunks; c += blockDim.x) {
    unsigned int lo = (unsigned int)(unsigned long long)(uintptr_t)(ldsDst + c * 8);
    unsigned long long ga = (unsigned long long)(uintptr_t)(src + c * 8);
    asm volatile("global_load_async_to_lds_b128 %0, %1, off"
                 :: "v"(lo), "v"(ga) : "memory");
  }
}

// Each wave drains its own async issues, then the block synchronizes.
// Every wave in the block must execute this exactly once.
DEV void stage_wait() {
  asm volatile("s_wait_asynccnt 0" ::: "memory");
  __syncthreads();
}

// One GEMM layer over a 32x128 LDS activation tile: acc += tile * W
template <int KC, int NT>
DEV void gemm_tile(const f16* tile, const f16* wf, v8f (&acc)[2][8], int lane) {
  int m = lane & 15, hf = lane >> 4;
#pragma unroll
  for (int kc = 0; kc < KC; ++kc) {
    v16h a0 = a_frag(tile + m * 128, kc * 32, hf);
    v16h a1 = a_frag(tile + (m + 16) * 128, kc * 32, hf);
#pragma unroll
    for (int nt = 0; nt < NT; ++nt) {
      v16h b = b_frag(wf, kc * NT + nt, lane);
      acc[0][nt] = wmma_f16(a0, b, acc[0][nt]);
      acc[1][nt] = wmma_f16(a1, b, acc[1][nt]);
    }
  }
}

// Bias + ReLU + convert to f16, C-layout -> row-major 32x128 LDS tile.
// C layout: element r of acc[s][nt] at lane l is row 16*s + r + 8*(l>>4),
// col 16*nt + (l&15).
DEV void bias_relu_to_tile(f16* tile, v8f (&acc)[2][8], const float* bias, int lane) {
  int col = lane & 15, rb = 8 * (lane >> 4);
#pragma unroll
  for (int nt = 0; nt < 8; ++nt) {
    float bv = bias[16 * nt + col];
#pragma unroll
    for (int s = 0; s < 2; ++s)
#pragma unroll
      for (int r = 0; r < 8; ++r) {
        float v = acc[s][nt][r] + bv;
        v = v > 0.f ? v : 0.f;
        tile[(16 * s + rb + r) * 128 + 16 * nt + col] = (f16)v;
      }
  }
}

// Layers 2 and 3 (both 128x128) from LDS-staged weights. acc out = pre-bias L3.
DEV void tail23(f16* tile, const f16* ldsW, v8f (&acc)[2][8],
                const float* b1, const float* b2, int lane) {
  bias_relu_to_tile(tile, acc, b1, lane);
  zero_acc(acc);
  gemm_tile<4, 8>(tile, ldsW, acc, lane);
  bias_relu_to_tile(tile, acc, b2, lane);
  zero_acc(acc);
  gemm_tile<4, 8>(tile, ldsW + 16384, acc, lane);
}

DEV void write_plain(float* out32, f16* outh, v8f (&acc)[2][8],
                     const float* b3, int row0, int lane) {
  int col = lane & 15, rb = 8 * (lane >> 4);
#pragma unroll
  for (int nt = 0; nt < 8; ++nt) {
    float bv = b3[16 * nt + col];
#pragma unroll
    for (int s = 0; s < 2; ++s)
#pragma unroll
      for (int r = 0; r < 8; ++r) {
        size_t idx = (size_t)(row0 + 16 * s + rb + r) * 128 + 16 * nt + col;
        float v = acc[s][nt][r] + bv;
        out32[idx] = v;
        outh[idx] = (f16)v;
      }
  }
}

DEV void write_residual(float* out32, f16* outh, v8f (&acc)[2][8],
                        const float* b3, int row0, int lane) {
  int col = lane & 15, rb = 8 * (lane >> 4);
#pragma unroll
  for (int nt = 0; nt < 8; ++nt) {
    float bv = b3[16 * nt + col];
#pragma unroll
    for (int s = 0; s < 2; ++s)
#pragma unroll
      for (int r = 0; r < 8; ++r) {
        size_t idx = (size_t)(row0 + 16 * s + rb + r) * 128 + 16 * nt + col;
        float nv = out32[idx] + acc[s][nt][r] + bv;
        out32[idx] = nv;
        outh[idx] = (f16)nv;
      }
  }
}

// ---------------------------------------------------------------------------
// Repack f32 row-major weights [Kreal x Nreal] into B fragments (f16, padded).
// ---------------------------------------------------------------------------
__global__ void repack_w_kernel(const float* __restrict__ W, int Kreal, int Nreal,
                                int KC, int NT, f16* __restrict__ out) {
  int t = blockIdx.x * blockDim.x + threadIdx.x;
  int total = KC * NT * 512;
  if (t >= total) return;
  int frag = t >> 9, r = t & 511;
  int lane = r >> 4, j = r & 15;
  int kc = frag / NT, nt = frag % NT;
  int K = 32 * kc + 8 * (lane >> 4) + 16 * (j >> 3) + (j & 7);
  int Nn = 16 * nt + (lane & 15);
  out[t] = (K < Kreal && Nn < Nreal) ? (f16)W[(size_t)K * Nreal + Nn] : (f16)0.f;
}

// ---------------------------------------------------------------------------
// Encoder: raw features (Kin<=12) -> 128, 3-layer MLP. K padded to 32.
// ---------------------------------------------------------------------------
__global__ void __launch_bounds__(128)
encode_mlp_kernel(const float* __restrict__ in, int Kin, int rows,
                  const f16* __restrict__ w1f, const f16* __restrict__ w23f,
                  const float* __restrict__ b1, const float* __restrict__ b2,
                  const float* __restrict__ b3,
                  float* __restrict__ out32, f16* __restrict__ outh) {
  extern __shared__ char smem[];
  f16* ldsW = (f16*)smem;
  stage_issue(ldsW, w23f, 32768);
  f16* tile = (f16*)(smem + 65536) + (threadIdx.x >> 5) * (32 * 128);
  int wave = blockIdx.x * (blockDim.x >> 5) + (threadIdx.x >> 5);
  int row0 = wave * 32;
  if (row0 >= rows) { stage_wait(); return; }
  int lane = threadIdx.x & 31, m = lane & 15, hf = lane >> 4;

  // layer 1 (K padded to 32) from raw f32 features, overlapped with staging
  AFrag ua0, ua1;
#pragma unroll
  for (int j = 0; j < 16; ++j) {
    int K = 8 * hf + 16 * (j >> 3) + (j & 7);
    ua0.v[j] = (K < Kin) ? (f16)in[(size_t)(row0 + m) * Kin + K] : (f16)0.f;
    ua1.v[j] = (K < Kin) ? (f16)in[(size_t)(row0 + 16 + m) * Kin + K] : (f16)0.f;
  }
  v8f acc[2][8];
  zero_acc(acc);
#pragma unroll
  for (int nt = 0; nt < 8; ++nt) {
    v16h b = b_frag(w1f, nt, lane);
    acc[0][nt] = wmma_f16(ua0.v, b, acc[0][nt]);
    acc[1][nt] = wmma_f16(ua1.v, b, acc[1][nt]);
  }
  stage_wait();
  tail23(tile, ldsW, acc, b1, b2, lane);
  write_plain(out32, outh, acc, b3, row0, lane);
}

// ---------------------------------------------------------------------------
// Edge processor: e += MLP([x[src], x[dst], e]) (K = 384)
// ---------------------------------------------------------------------------
__global__ void __launch_bounds__(128)
edge_update_kernel(const f16* __restrict__ xh, const int* __restrict__ src,
                   const int* __restrict__ dst, float* __restrict__ e32,
                   f16* __restrict__ eh, int E,
                   const f16* __restrict__ w1f, const f16* __restrict__ w23f,
                   const float* __restrict__ b1, const float* __restrict__ b2,
                   const float* __restrict__ b3) {
  extern __shared__ char smem[];
  f16* ldsW = (f16*)smem;
  stage_issue(ldsW, w23f, 32768);
  f16* tile = (f16*)(smem + 65536) + (threadIdx.x >> 5) * (32 * 128);
  int wave = blockIdx.x * (blockDim.x >> 5) + (threadIdx.x >> 5);
  int row0 = wave * 32;
  if (row0 >= E) { stage_wait(); return; }
  int lane = threadIdx.x & 31, m = lane & 15, hf = lane >> 4;

  // layer 1: gathered concat [x[src], x[dst], e], overlapped with staging
  int e0 = row0 + m, e1 = row0 + 16 + m;
  const f16* A0[3] = { xh + (size_t)src[e0] * 128, xh + (size_t)dst[e0] * 128,
                       eh + (size_t)e0 * 128 };
  const f16* A1[3] = { xh + (size_t)src[e1] * 128, xh + (size_t)dst[e1] * 128,
                       eh + (size_t)e1 * 128 };
  v8f acc[2][8];
  zero_acc(acc);
#pragma unroll
  for (int kc = 0; kc < 12; ++kc) {
    int kk = (kc & 3) * 32;
    v16h a0 = a_frag(A0[kc >> 2], kk, hf);
    v16h a1 = a_frag(A1[kc >> 2], kk, hf);
#pragma unroll
    for (int nt = 0; nt < 8; ++nt) {
      v16h b = b_frag(w1f, kc * 8 + nt, lane);
      acc[0][nt] = wmma_f16(a0, b, acc[0][nt]);
      acc[1][nt] = wmma_f16(a1, b, acc[1][nt]);
    }
  }
  stage_wait();
  tail23(tile, ldsW, acc, b1, b2, lane);
  write_residual(e32, eh, acc, b3, row0, lane);
}

// ---------------------------------------------------------------------------
// Node processor: x += MLP([x, agg_m, agg_c]) (K = 384)
// ---------------------------------------------------------------------------
__global__ void __launch_bounds__(128)
node_update_kernel(f16* __restrict__ xh, const f16* __restrict__ aggmh,
                   const f16* __restrict__ aggch, float* __restrict__ x32, int N,
                   const f16* __restrict__ w1f, const f16* __restrict__ w23f,
                   const float* __restrict__ b1, const float* __restrict__ b2,
                   const float* __restrict__ b3) {
  extern __shared__ char smem[];
  f16* ldsW = (f16*)smem;
  stage_issue(ldsW, w23f, 32768);
  f16* tile = (f16*)(smem + 65536) + (threadIdx.x >> 5) * (32 * 128);
  int wave = blockIdx.x * (blockDim.x >> 5) + (threadIdx.x >> 5);
  int row0 = wave * 32;
  if (row0 >= N) { stage_wait(); return; }
  int lane = threadIdx.x & 31, m = lane & 15, hf = lane >> 4;

  size_t r0 = (size_t)(row0 + m) * 128, r1 = (size_t)(row0 + 16 + m) * 128;
  const f16* A0[3] = { xh + r0, aggmh + r0, aggch + r0 };
  const f16* A1[3] = { xh + r1, aggmh + r1, aggch + r1 };
  v8f acc[2][8];
  zero_acc(acc);
#pragma unroll
  for (int kc = 0; kc < 12; ++kc) {
    int kk = (kc & 3) * 32;
    v16h a0 = a_frag(A0[kc >> 2], kk, hf);
    v16h a1 = a_frag(A1[kc >> 2], kk, hf);
#pragma unroll
    for (int nt = 0; nt < 8; ++nt) {
      v16h b = b_frag(w1f, kc * 8 + nt, lane);
      acc[0][nt] = wmma_f16(a0, b, acc[0][nt]);
      acc[1][nt] = wmma_f16(a1, b, acc[1][nt]);
    }
  }
  stage_wait();
  tail23(tile, ldsW, acc, b1, b2, lane);
  write_residual(x32, xh, acc, b3, row0, lane);
}

// ---------------------------------------------------------------------------
// Decoder: x -> 128 -> 128 -> 3
// ---------------------------------------------------------------------------
__global__ void __launch_bounds__(128)
decode_kernel(const f16* __restrict__ xh, int rows,
              const f16* __restrict__ w1f, const f16* __restrict__ w23f,
              const float* __restrict__ b1, const float* __restrict__ b2,
              const float* __restrict__ b3, float* __restrict__ out) {
  extern __shared__ char smem[];
  f16* ldsW = (f16*)smem;
  stage_issue(ldsW, w23f, 18432);  // W2 (16384) + W3 (2048)
  f16* tile = (f16*)(smem + 36864) + (threadIdx.x >> 5) * (32 * 128);
  int wave = blockIdx.x * (blockDim.x >> 5) + (threadIdx.x >> 5);
  int row0 = wave * 32;
  if (row0 >= rows) { stage_wait(); return; }
  int lane = threadIdx.x & 31, m = lane & 15, hf = lane >> 4;

  const f16* r0 = xh + (size_t)(row0 + m) * 128;
  const f16* r1 = xh + (size_t)(row0 + 16 + m) * 128;
  v8f acc[2][8];
  zero_acc(acc);
#pragma unroll
  for (int kc = 0; kc < 4; ++kc) {
    v16h a0 = a_frag(r0, kc * 32, hf);
    v16h a1 = a_frag(r1, kc * 32, hf);
#pragma unroll
    for (int nt = 0; nt < 8; ++nt) {
      v16h b = b_frag(w1f, kc * 8 + nt, lane);
      acc[0][nt] = wmma_f16(a0, b, acc[0][nt]);
      acc[1][nt] = wmma_f16(a1, b, acc[1][nt]);
    }
  }
  stage_wait();
  bias_relu_to_tile(tile, acc, b1, lane);
  zero_acc(acc);
  gemm_tile<4, 8>(tile, ldsW, acc, lane);
  bias_relu_to_tile(tile, acc, b2, lane);
  zero_acc(acc);
  gemm_tile<4, 1>(tile, ldsW + 16384, acc, lane);

  int col = lane & 15, rb = 8 * hf;
  if (col < 3) {
    float bv = b3[col];
#pragma unroll
    for (int s = 0; s < 2; ++s)
#pragma unroll
      for (int r = 0; r < 8; ++r)
        out[(size_t)(row0 + 16 * s + rb + r) * 3 + col] = acc[s][0][r] + bv;
  }
}

// ---------------------------------------------------------------------------
// Aggregation helpers
// ---------------------------------------------------------------------------
__global__ void deg_count_kernel(const int* __restrict__ dst, int E,
                                 float* __restrict__ deg) {
  int t = blockIdx.x * blockDim.x + threadIdx.x;
  if (t < E) unsafeAtomicAdd(&deg[dst[t]], 1.0f);
}

__global__ void inv_deg_kernel(float* __restrict__ d, int N) {
  int t = blockIdx.x * blockDim.x + threadIdx.x;
  if (t < N) d[t] = 1.0f / fmaxf(d[t], 1.0f);
}

__global__ void scatter_add_kernel(const float* __restrict__ e,
                                   const int* __restrict__ dst, int E,
                                   float* __restrict__ agg) {
  int t = blockIdx.x * blockDim.x + threadIdx.x;
  int ed = t >> 2;
  if (ed >= E) return;
  int cg = (t & 3) * 32;
  int dn = dst[ed];
  const float* ep = e + (size_t)ed * 128 + cg;
  float* ap = agg + (size_t)dn * 128 + cg;
#pragma unroll
  for (int i = 0; i < 32; ++i) unsafeAtomicAdd(ap + i, ep[i]);
}

__global__ void normalize_kernel(float* __restrict__ agg, f16* __restrict__ aggh,
                                 const float* __restrict__ inv, int total) {
  int t = blockIdx.x * blockDim.x + threadIdx.x;
  if (t >= total) return;
  float v = agg[t] * inv[t >> 7];
  agg[t] = v;
  aggh[t] = (f16)v;
}

// ---------------------------------------------------------------------------
// Host orchestration
// ---------------------------------------------------------------------------
extern "C" void kernel_launch(void* const* d_in, const int* in_sizes, int n_in,
                              void* d_out, int out_size, void* d_ws, size_t ws_size,
                              hipStream_t stream) {
  const float* node_x = (const float*)d_in[0];
  const float* mesh_ea = (const float*)d_in[1];
  const float* cont_ea = (const float*)d_in[2];
  const float* P[36];
  for (int i = 0; i < 36; ++i) P[i] = (const float*)d_in[3 + i];
  const int* mei = (const int*)d_in[39];
  const int* cei = (const int*)d_in[40];

  const int N = in_sizes[0] / 12;
  const int EM = in_sizes[39] / 2;
  const int EC = in_sizes[40] / 2;
  const int* msrc = mei;
  const int* mdst = mei + EM;
  const int* csrc = cei;
  const int* cdst = cei + EC;

  // bump allocator over workspace
  char* ws = (char*)d_ws;
  size_t off = 0;
  auto alloc = [&](size_t bytes) -> void* {
    void* p = ws + off;
    off = (off + bytes + 255) & ~(size_t)255;
    return p;
  };

  float* x32 = (float*)alloc((size_t)N * 128 * 4);
  f16* xh = (f16*)alloc((size_t)N * 128 * 2);
  float* me32 = (float*)alloc((size_t)EM * 128 * 4);
  f16* meh = (f16*)alloc((size_t)EM * 128 * 2);
  float* ce32 = (float*)alloc((size_t)EC * 128 * 4);
  f16* ceh = (f16*)alloc((size_t)EC * 128 * 2);
  float* aggm = (float*)alloc((size_t)N * 128 * 4);
  float* aggc = (float*)alloc((size_t)N * 128 * 4);
  f16* aggmh = (f16*)alloc((size_t)N * 128 * 2);
  f16* aggch = (f16*)alloc((size_t)N * 128 * 2);
  float* invm = (float*)alloc((size_t)N * 4);
  float* invc = (float*)alloc((size_t)N * 4);
  f16* nef = (f16*)alloc(36864 * 2);
  f16* mef = (f16*)alloc(36864 * 2);
  f16* cef = (f16*)alloc(36864 * 2);
  f16* pef = (f16*)alloc(81920 * 2);
  f16* pnf = (f16*)alloc(81920 * 2);
  f16* def = (f16*)alloc(34816 * 2);

  auto repack = [&](const float* W, int Kr, int Nr, int KC, int NT, f16* o) {
    int total = KC * NT * 512;
    repack_w_kernel<<<(total + 255) / 256, 256, 0, stream>>>(W, Kr, Nr, KC, NT, o);
  };
  // encoders (K padded to 32)
  repack(P[0], 12, 128, 1, 8, nef);
  repack(P[2], 128, 128, 4, 8, nef + 4096);
  repack(P[4], 128, 128, 4, 8, nef + 20480);
  repack(P[6], 7, 128, 1, 8, mef);
  repack(P[8], 128, 128, 4, 8, mef + 4096);
  repack(P[10], 128, 128, 4, 8, mef + 20480);
  repack(P[12], 4, 128, 1, 8, cef);
  repack(P[14], 128, 128, 4, 8, cef + 4096);
  repack(P[16], 128, 128, 4, 8, cef + 20480);
  // processors (K = 384)
  repack(P[18], 384, 128, 12, 8, pef);
  repack(P[20], 128, 128, 4, 8, pef + 49152);
  repack(P[22], 128, 128, 4, 8, pef + 65536);
  repack(P[24], 384, 128, 12, 8, pnf);
  repack(P[26], 128, 128, 4, 8, pnf + 49152);
  repack(P[28], 128, 128, 4, 8, pnf + 65536);
  // decoder (final N = 3 padded to 16)
  repack(P[30], 128, 128, 4, 8, def);
  repack(P[32], 128, 128, 4, 8, def + 16384);
  repack(P[34], 128, 3, 4, 1, def + 32768);

  auto blocksFor = [](int rows) { return (rows / 32 + 3) / 4; };
  const size_t SH_PROC = 65536 + 32768;   // W23 + 4 activation tiles
  const size_t SH_DEC = 36864 + 32768;

  // degree normalization factors (loop-invariant)
  hipMemsetAsync(invm, 0, (size_t)N * 4, stream);
  hipMemsetAsync(invc, 0, (size_t)N * 4, stream);
  deg_count_kernel<<<(EM + 255) / 256, 256, 0, stream>>>(mdst, EM, invm);
  deg_count_kernel<<<(EC + 255) / 256, 256, 0, stream>>>(cdst, EC, invc);
  inv_deg_kernel<<<(N + 255) / 256, 256, 0, stream>>>(invm, N);
  inv_deg_kernel<<<(N + 255) / 256, 256, 0, stream>>>(invc, N);

  // encode
  encode_mlp_kernel<<<blocksFor(N), 128, SH_PROC, stream>>>(
      node_x, 12, N, nef, nef + 4096, P[1], P[3], P[5], x32, xh);
  encode_mlp_kernel<<<blocksFor(EM), 128, SH_PROC, stream>>>(
      mesh_ea, 7, EM, mef, mef + 4096, P[7], P[9], P[11], me32, meh);
  encode_mlp_kernel<<<blocksFor(EC), 128, SH_PROC, stream>>>(
      cont_ea, 4, EC, cef, cef + 4096, P[13], P[15], P[17], ce32, ceh);

  // process: 10 message-passing steps
  for (int step = 0; step < 10; ++step) {
    edge_update_kernel<<<blocksFor(EM), 128, SH_PROC, stream>>>(
        xh, msrc, mdst, me32, meh, EM, pef, pef + 49152, P[19], P[21], P[23]);
    edge_update_kernel<<<blocksFor(EC), 128, SH_PROC, stream>>>(
        xh, csrc, cdst, ce32, ceh, EC, pef, pef + 49152, P[19], P[21], P[23]);
    hipMemsetAsync(aggm, 0, (size_t)N * 128 * 4, stream);
    hipMemsetAsync(aggc, 0, (size_t)N * 128 * 4, stream);
    scatter_add_kernel<<<(EM * 4 + 127) / 128, 128, 0, stream>>>(me32, mdst, EM, aggm);
    scatter_add_kernel<<<(EC * 4 + 127) / 128, 128, 0, stream>>>(ce32, cdst, EC, aggc);
    normalize_kernel<<<(N * 128 + 255) / 256, 256, 0, stream>>>(aggm, aggmh, invm, N * 128);
    normalize_kernel<<<(N * 128 + 255) / 256, 256, 0, stream>>>(aggc, aggch, invc, N * 128);
    node_update_kernel<<<blocksFor(N), 128, SH_PROC, stream>>>(
        xh, aggmh, aggch, x32, N, pnf, pnf + 49152, P[25], P[27], P[29]);
  }

  // decode
  decode_kernel<<<blocksFor(N), 128, SH_DEC, stream>>>(
      xh, N, def, def + 16384, P[31], P[33], P[35], (float*)d_out);
}
